// ContrastiveLoss_45775761441010
// MI455X (gfx1250) — matrix operand order repaired and checked by reading, not compiled, not run on hardware
//
#include <hip/hip_runtime.h>
#include <hip/hip_bf16.h>
#include <stdint.h>

#define B_SZ   512
#define D_SZ   2048
#define PM_SZ  8
#define PH_SZ  16
#define NM_SZ  64
#define NH_SZ  64
#define NVEC   (PM_SZ + PH_SZ + NM_SZ + NH_SZ)   // 152
#define INV_T  (1.0f / 0.07f)
#define POS_MARGIN 0.1f
#define NEG_MARGIN 0.1f

typedef __bf16 bf16_t;
typedef __attribute__((ext_vector_type(4)))  float  v4f;
typedef __attribute__((ext_vector_type(8)))  float  v8f;
typedef __attribute__((ext_vector_type(8)))  __bf16 v8bf;
typedef __attribute__((ext_vector_type(16))) __bf16 v16bf;

__device__ __forceinline__ bf16_t f2bf(float f) {
    union { float f; unsigned u; } v; v.f = f;
    unsigned r = v.u + 0x7FFFu + ((v.u >> 16) & 1u);   // round-to-nearest-even
    unsigned short h = (unsigned short)(r >> 16);
    bf16_t out;
    __builtin_memcpy(&out, &h, 2);
    return out;
}

// ---------------------------------------------------------------------------
// Kernel 1: L2-normalize q rows and pf = pos_main[:,0] rows into bf16, and
// record 1/||q_b||. grid = 1024 blocks (0..511 -> q, 512..1023 -> pf), 256 thr.
// Thread t owns elements [8t, 8t+8): two float4 loads, one 16B bf16 store.
// ---------------------------------------------------------------------------
__global__ void normalize_kernel(const float* __restrict__ q,
                                 const float* __restrict__ pm,
                                 bf16_t* __restrict__ qn,
                                 bf16_t* __restrict__ pfn,
                                 float* __restrict__ inv_q) {
    __shared__ float red[8];
    const int idx = blockIdx.x;
    const bool isQ = idx < B_SZ;
    const int b = isQ ? idx : idx - B_SZ;
    const float* src = isQ ? (q + (size_t)b * D_SZ)
                           : (pm + (size_t)b * PM_SZ * D_SZ);   // row 0 of pos_main[b]
    bf16_t* dst = isQ ? (qn + (size_t)b * D_SZ) : (pfn + (size_t)b * D_SZ);

    const int t = threadIdx.x;
    const v4f* s4 = reinterpret_cast<const v4f*>(src);
    v4f a = s4[2 * t];
    v4f c = s4[2 * t + 1];
    float local = a.x * a.x + a.y * a.y + a.z * a.z + a.w * a.w
                + c.x * c.x + c.y * c.y + c.z * c.z + c.w * c.w;

    const int lane = t & 31;
    const int wave = t >> 5;
#pragma unroll
    for (int m = 16; m >= 1; m >>= 1) local += __shfl_xor(local, m, 32);
    if (lane == 0) red[wave] = local;
    __syncthreads();
    if (t == 0) {
        float tot = 0.f;
#pragma unroll
        for (int i = 0; i < 8; ++i) tot += red[i];
        red[0] = 1.f / fmaxf(sqrtf(tot), 1e-8f);
    }
    __syncthreads();
    const float inv = red[0];

    v8bf o;
    o[0] = f2bf(a.x * inv); o[1] = f2bf(a.y * inv);
    o[2] = f2bf(a.z * inv); o[3] = f2bf(a.w * inv);
    o[4] = f2bf(c.x * inv); o[5] = f2bf(c.y * inv);
    o[6] = f2bf(c.z * inv); o[7] = f2bf(c.w * inv);
    *reinterpret_cast<v8bf*>(dst + 8 * t) = o;      // 16B aligned
    if (isQ && t == 0) inv_q[b] = inv;
}

// ---------------------------------------------------------------------------
// Kernel 2: per-batch sims (152 dots of length 2048 per row), anchor/threshold,
// pos_score and the nm/nh part of neg_score. One block (8 waves) per batch row.
// float4 non-temporal global loads (single-use 600MB stream) + float4 LDS reads.
// ---------------------------------------------------------------------------
__global__ void sims_kernel(const float* __restrict__ q,
                            const float* __restrict__ pm,
                            const float* __restrict__ ph,
                            const float* __restrict__ nm,
                            const float* __restrict__ nh,
                            const float* __restrict__ inv_q,
                            float* __restrict__ pos_score,
                            float* __restrict__ neg_score,
                            float* __restrict__ thr_out) {
    __shared__ v4f qs[D_SZ / 4];
    __shared__ float sims[NVEC];
    __shared__ float s_pos, s_neg;

    const int b = blockIdx.x;
    const v4f* qrow = reinterpret_cast<const v4f*>(q + (size_t)b * D_SZ);
    qs[threadIdx.x]       = qrow[threadIdx.x];
    qs[threadIdx.x + 256] = qrow[threadIdx.x + 256];
    if (threadIdx.x == 0) { s_pos = 0.f; s_neg = 0.f; }
    const float invq = inv_q[b];
    __syncthreads();

    const int wave = threadIdx.x >> 5;
    const int lane = threadIdx.x & 31;
    for (int v = wave; v < NVEC; v += 8) {
        const float* src;
        if (v < PM_SZ)                src = pm + ((size_t)b * PM_SZ + v) * D_SZ;
        else if (v < PM_SZ + PH_SZ)   src = ph + ((size_t)b * PH_SZ + (v - PM_SZ)) * D_SZ;
        else if (v < PM_SZ + PH_SZ + NM_SZ)
                                      src = nm + ((size_t)b * NM_SZ + (v - PM_SZ - PH_SZ)) * D_SZ;
        else                          src = nh + ((size_t)b * NH_SZ + (v - PM_SZ - PH_SZ - NM_SZ)) * D_SZ;
        const v4f* src4 = reinterpret_cast<const v4f*>(src);
        float dot = 0.f, ss = 0.f;
#pragma unroll 4
        for (int k = lane; k < D_SZ / 4; k += 32) {
            v4f x  = __builtin_nontemporal_load(&src4[k]);
            v4f qv = qs[k];
            dot = fmaf(qv.x, x.x, dot); dot = fmaf(qv.y, x.y, dot);
            dot = fmaf(qv.z, x.z, dot); dot = fmaf(qv.w, x.w, dot);
            ss  = fmaf(x.x, x.x, ss);   ss  = fmaf(x.y, x.y, ss);
            ss  = fmaf(x.z, x.z, ss);   ss  = fmaf(x.w, x.w, ss);
        }
#pragma unroll
        for (int m = 16; m >= 1; m >>= 1) {
            dot += __shfl_xor(dot, m, 32);
            ss  += __shfl_xor(ss,  m, 32);
        }
        if (lane == 0) sims[v] = dot * invq / fmaxf(sqrtf(ss), 1e-8f);
    }
    __syncthreads();

    const float anchor = sims[0];
    const float thr = anchor + NEG_MARGIN;
    if (threadIdx.x < NVEC) {
        const float s = sims[threadIdx.x];
        const float e = __expf(s * INV_T);
        if (threadIdx.x < PM_SZ) {
            atomicAdd(&s_pos, e);
        } else if (threadIdx.x < PM_SZ + PH_SZ) {
            if (s >= anchor - POS_MARGIN) atomicAdd(&s_pos, e);
        } else {
            if (s <= thr) atomicAdd(&s_neg, e);
        }
    }
    __syncthreads();
    if (threadIdx.x == 0) {
        pos_score[b] = s_pos;
        neg_score[b] = s_neg;     // cross-batch kernel atomics on top of this
        thr_out[b]   = thr;
    }
}

// ---------------------------------------------------------------------------
// WMMA fragment loaders for v_wmma_f32_16x16x32_bf16 (ISA §7.12.2 layouts).
// Rows of the source matrices are contiguous (ld = 2048 bf16, 4KB aligned).
// ---------------------------------------------------------------------------
__device__ __forceinline__ v16bf load_frag_a(const bf16_t* __restrict__ base,
                                             int rowBase, int kBase, int lane) {
    // A (16x32, MxK): lanes 0-15 row M=lane, elems 0..7 = K 0..7, 8..15 = K 16..23
    //                 lanes 16-31 row M=lane-16, elems 0..7 = K 8..15, 8..15 = K 24..31
    const int m  = lane & 15;
    const int kh = (lane >> 4) << 3;                   // 0 or 8
    const bf16_t* p = base + (size_t)(rowBase + m) * D_SZ + kBase;
    v8bf x0 = *reinterpret_cast<const v8bf*>(p + kh);        // 16B aligned
    v8bf x1 = *reinterpret_cast<const v8bf*>(p + 16 + kh);
    return __builtin_shufflevector(x0, x1, 0,1,2,3,4,5,6,7,8,9,10,11,12,13,14,15);
}

__device__ __forceinline__ v16bf load_frag_bt(const bf16_t* __restrict__ base,
                                              int colBase, int kBase, int lane) {
    // B (32x16, KxN) with B = Y^T: lane n holds column N=lane&15 = row n of Y.
    // lanes 0-15: elems 0..7 = K 0..7, 8..15 = K 16..23; lanes 16-31: K 8..15, K 24..31.
    const int n  = lane & 15;
    const int kb = (lane >> 4) << 3;                   // 0 or 8
    const bf16_t* p = base + (size_t)(colBase + n) * D_SZ + kBase + kb;
    v8bf x0 = *reinterpret_cast<const v8bf*>(p);
    v8bf x1 = *reinterpret_cast<const v8bf*>(p + 16);
    return __builtin_shufflevector(x0, x1, 0,1,2,3,4,5,6,7,8,9,10,11,12,13,14,15);
}

// ---------------------------------------------------------------------------
// Kernel 3: one wave per 16x16 output tile (i,j). Computes the tile of all
// three products qq = qn@qn^T, pp = pfn@pfn^T, qp = qn@pfn^T over K=2048 with
// v_wmma_f32_16x16x32_bf16, then fuses the masked exp row-reduction and
// atomically accumulates into neg_score. grid = (32,32), block = 32 (wave32,
// EXEC all-ones as WMMA requires).
// ---------------------------------------------------------------------------
__global__ void gemm_mask_kernel(const bf16_t* __restrict__ qn,
                                 const bf16_t* __restrict__ pfn,
                                 const float* __restrict__ thr,
                                 float* __restrict__ neg_score) {
    const int lane = threadIdx.x;          // 32-thread block
    const int colBase = blockIdx.x * 16;   // j tile
    const int rowBase = blockIdx.y * 16;   // i tile

    v8f acc_qq = {}, acc_pp = {}, acc_qp = {};
#pragma unroll 4
    for (int kb = 0; kb < D_SZ; kb += 32) {
        v16bf aq = load_frag_a (qn,  rowBase, kb, lane);
        v16bf ap = load_frag_a (pfn, rowBase, kb, lane);
        v16bf bq = load_frag_bt(qn,  colBase, kb, lane);
        v16bf bp = load_frag_bt(pfn, colBase, kb, lane);
        acc_qq = __builtin_amdgcn_wmma_f32_16x16x32_bf16(false, aq, false, bq,
                                                         (short)0, acc_qq, false, false);
        acc_pp = __builtin_amdgcn_wmma_f32_16x16x32_bf16(false, ap, false, bp,
                                                         (short)0, acc_pp, false, false);
        acc_qp = __builtin_amdgcn_wmma_f32_16x16x32_bf16(false, aq, false, bp,
                                                         (short)0, acc_qp, false, false);
    }

    // C/D layout: VGPR r, lane L -> M = r + 8*(L>>4), N = L&15.
    const int nloc = lane & 15;
    const int half = lane >> 4;
    const int gj = colBase + nloc;
#pragma unroll
    for (int r = 0; r < 8; ++r) {
        const int gi = rowBase + r + 8 * half;
        const float t = thr[gi];
        float s = 0.f;
        if (gi != gj) {
            const float vqq = acc_qq[r];
            const float vpp = acc_pp[r];
            const float vqp = acc_qp[r];
            if (vqq <= t) s += __expf(vqq * INV_T);
            if (vpp <= t) s += __expf(vpp * INV_T);
            if (vqp <= t) s += __expf(vqp * INV_T);
        }
        // reduce across the 16 lanes of this half-wave (same row gi)
#pragma unroll
        for (int m = 8; m >= 1; m >>= 1) s += __shfl_xor(s, m, 32);
        if (nloc == 0) atomicAdd(&neg_score[gi], s);
    }
}

// ---------------------------------------------------------------------------
// Kernel 4: loss = mean_b( -log(pos/(pos+neg+1e-8)) ). One block, 512 threads.
// ---------------------------------------------------------------------------
__global__ void finalize_kernel(const float* __restrict__ pos_score,
                                const float* __restrict__ neg_score,
                                float* __restrict__ out) {
    __shared__ float red[16];
    const int t = threadIdx.x;
    const float pos = pos_score[t];
    const float neg = neg_score[t];
    float l = -logf(pos / (pos + neg + 1e-8f));
    const int lane = t & 31;
    const int wave = t >> 5;
#pragma unroll
    for (int m = 16; m >= 1; m >>= 1) l += __shfl_xor(l, m, 32);
    if (lane == 0) red[wave] = l;
    __syncthreads();
    if (t == 0) {
        float s = 0.f;
#pragma unroll
        for (int i = 0; i < 16; ++i) s += red[i];
        out[0] = s / (float)B_SZ;
    }
}

// ---------------------------------------------------------------------------
extern "C" void kernel_launch(void* const* d_in, const int* in_sizes, int n_in,
                              void* d_out, int out_size, void* d_ws, size_t ws_size,
                              hipStream_t stream) {
    const float* q  = (const float*)d_in[0];
    const float* pm = (const float*)d_in[1];
    const float* ph = (const float*)d_in[2];
    const float* nm = (const float*)d_in[3];
    const float* nh = (const float*)d_in[4];
    float* out = (float*)d_out;

    char* ws = (char*)d_ws;
    bf16_t* qn    = (bf16_t*)ws;                              // 512*2048*2 = 2 MB
    bf16_t* pfn   = (bf16_t*)(ws + (size_t)2 * 1024 * 1024);  // 2 MB
    float*  inv_q = (float*)(ws + (size_t)4 * 1024 * 1024);   // 512 f
    float*  thr   = inv_q + B_SZ;                             // 512 f
    float*  pos_s = thr + B_SZ;                               // 512 f
    float*  neg_s = pos_s + B_SZ;                             // 512 f

    normalize_kernel<<<dim3(2 * B_SZ), dim3(256), 0, stream>>>(q, pm, qn, pfn, inv_q);
    sims_kernel<<<dim3(B_SZ), dim3(256), 0, stream>>>(q, pm, ph, nm, nh, inv_q,
                                                      pos_s, neg_s, thr);
    gemm_mask_kernel<<<dim3(B_SZ / 16, B_SZ / 16), dim3(32), 0, stream>>>(qn, pfn,
                                                                          thr, neg_s);
    finalize_kernel<<<dim3(1), dim3(512), 0, stream>>>(pos_s, neg_s, out);
}
